// ViTImageCaptioningModel_47682726920350
// MI455X (gfx1250) — matrix operand order repaired and checked by reading, not compiled, not run on hardware
//
#include <hip/hip_runtime.h>
#include <hip/hip_bf16.h>

// ---------------------------------------------------------------------------
// ViT image-captioning decoder for MI455X (gfx1250, wave32, WMMA).
//
// FLOP budget: logits GEMM (2432x512 @ 512x49408) = 123 GFLOP dominates.
// bf16 WMMA (v_wmma_f32_16x16x32_bf16, f32 accumulate) for all GEMMs;
// LayerNorm / softmax / residuals / bias in fp32 VALU.
//
// GEMM data movement: weights pre-transposed to bf16 [N,K]; each workgroup's
// 64-column B strip is one contiguous 64KB block, staged into LDS ONCE via
// CDNA5 async copies (global_load_async_to_lds_b128 + s_wait_asynccnt), then
// shared by all 8 waves -> 8x reduction in B-side VMEM traffic. LDS rows are
// padded (520 bf16 = 1040B) so 16-lane ds_load_b128 fragments are
// bank-conflict-free (bank rotation of 4 dwords per row).
// ---------------------------------------------------------------------------

typedef __bf16 bf16_t;
typedef __bf16 v16bf __attribute__((ext_vector_type(16)));
typedef __bf16 v8bf  __attribute__((ext_vector_type(8)));
typedef float  v8f   __attribute__((ext_vector_type(8)));

#define IMG_LEN 50
#define TXT_LEN 76
#define SEQ     126
#define BATCH   32
#define DMODEL  512
#define DIMG    768
#define VOCAB   49408
#define ROWS    (BATCH * SEQ)     // 4032 = 252 * 16
#define TROWS   (BATCH * TXT_LEN) // 2432 = 152 * 16
#define KDIM    512               // GEMM K (all GEMMs share K = DMODEL)
#define BSTRIDE 520               // padded LDS row stride in bf16 elements

// ---------------------------------------------------------------------------
// Weight transpose + fp32 -> bf16:  W[K,N] -> Wt[N,K]
// ---------------------------------------------------------------------------
__global__ __launch_bounds__(256) void transpose_to_bf16(
    const float* __restrict__ W, bf16_t* __restrict__ Wt, int K, int N)
{
    size_t idx = (size_t)blockIdx.x * 256 + threadIdx.x;
    if (idx >= (size_t)K * N) return;
    int k = (int)(idx / N);
    int n = (int)(idx % N);
    Wt[(size_t)n * K + k] = (bf16_t)W[idx];
}

// ---------------------------------------------------------------------------
// Embed: rows 0..49 = image @ ff0_w + ff0_b ; rows 50..125 = token + pos.
// ---------------------------------------------------------------------------
__global__ __launch_bounds__(256) void embed_kernel(
    const float* __restrict__ img, const int* __restrict__ ids,
    const float* __restrict__ tok, const float* __restrict__ pos,
    const float* __restrict__ ff0w, const float* __restrict__ ff0b,
    float* __restrict__ out)
{
    int idx = blockIdx.x * 256 + threadIdx.x;   // ROWS*DMODEL threads
    if (idx >= ROWS * DMODEL) return;
    int row = idx >> 9;          // /512
    int d   = idx & (DMODEL - 1);
    int b   = row / SEQ;
    int s   = row - b * SEQ;
    float val;
    if (s < IMG_LEN) {
        const float* ip = img + ((size_t)b * IMG_LEN + s) * DIMG;
        float acc = ff0b[d];
        #pragma unroll 4
        for (int k = 0; k < DIMG; ++k)
            acc = fmaf(ip[k], ff0w[(size_t)k * DMODEL + d], acc);
        val = acc;
    } else {
        int t  = s - IMG_LEN;
        int id = ids[b * TXT_LEN + t];
        val = tok[(size_t)id * DMODEL + d] + pos[(size_t)t * DMODEL + d];
    }
    out[idx] = val;
}

// ---------------------------------------------------------------------------
// LayerNorm over D=512; optional residual. One wave32 per row, 16 floats per
// lane, shuffle reduction. Writes fp32 X and bf16 Xb (fused convert).
// ---------------------------------------------------------------------------
__global__ __launch_bounds__(256) void ln_kernel(
    const float* __restrict__ src, const float* __restrict__ resid,
    const float* __restrict__ g, const float* __restrict__ bparam,
    float* __restrict__ X, bf16_t* __restrict__ Xb, int nrows)
{
    int wave = threadIdx.x >> 5;
    int lane = threadIdx.x & 31;
    int row  = blockIdx.x * 8 + wave;
    if (row >= nrows) return;

    const size_t base = (size_t)row * DMODEL + lane * 16;
    float v[16];
    #pragma unroll
    for (int i = 0; i < 16; ++i) v[i] = src[base + i];
    if (resid) {
        #pragma unroll
        for (int i = 0; i < 16; ++i) v[i] += resid[base + i];
    }
    float sum = 0.f;
    #pragma unroll
    for (int i = 0; i < 16; ++i) sum += v[i];
    #pragma unroll
    for (int off = 16; off >= 1; off >>= 1) sum += __shfl_xor(sum, off, 32);
    const float mean = sum * (1.0f / DMODEL);

    float var = 0.f;
    #pragma unroll
    for (int i = 0; i < 16; ++i) { float dd = v[i] - mean; var += dd * dd; }
    #pragma unroll
    for (int off = 16; off >= 1; off >>= 1) var += __shfl_xor(var, off, 32);
    const float inv = rsqrtf(var * (1.0f / DMODEL) + 1e-5f);

    #pragma unroll
    for (int i = 0; i < 16; ++i) {
        int c = lane * 16 + i;
        float y = (v[i] - mean) * inv * g[c] + bparam[c];
        X[base + i]  = y;
        Xb[base + i] = (bf16_t)y;
    }
}

// ---------------------------------------------------------------------------
// bf16 WMMA GEMM:  C[M,N] = A[M,K] * B[K,N] + bias,  B pre-transposed as
// Bt[N,K] bf16, K = 512. Block = 256 threads = 8 waves.
//
// Phase 1: all 256 threads async-copy the contiguous 64KB B strip
//          Bt[nbase..nbase+63][0..511] into padded LDS (ASYNCcnt path).
// Phase 2: wave w computes M-tile (blockIdx.y*8+w) x 64 columns:
//          A fragments from global (2x16B per lane), B fragments from LDS
//          (2x16B ds_load per lane, conflict-free via BSTRIDE padding),
//          4 WMMA accumulators.
//
// Fragment layouts used by this kernel (wave32; A/B k order is contiguous, k = 16h+s):
//   A 16x32 : lane m=(lane&15), h=lane>>4; elems 0..15 = A[m][k0+16h+0..15]
//   B 32x16 : lane n=(lane&15), h=lane>>4; elems 0..15 = Bt[n][k0+16h+0..15]
//   C 16x16 : vgpr r, half h -> row 8h+r, col = lane&15
// ---------------------------------------------------------------------------
__global__ __launch_bounds__(256) void gemm_bf16_wmma(
    const bf16_t* __restrict__ A, const bf16_t* __restrict__ Bt,
    const float* __restrict__ bias, float* __restrict__ C,
    int Mtiles, int N)
{
    __shared__ __align__(16) bf16_t Bs[64 * BSTRIDE];   // 66,560 B

    const int tid   = threadIdx.x;
    const int nbase = blockIdx.x * 64;

    // ---- Phase 1: async-stage the 64-col B strip into LDS -----------------
    {
        const bf16_t* gsrc = Bt + (size_t)nbase * KDIM;      // contiguous 64KB
        const unsigned lbase = (unsigned)(unsigned long long)(&Bs[0]);
        #pragma unroll
        for (int j = 0; j < 16; ++j) {
            int chunk = j * 256 + tid;           // 4096 chunks of 16B
            int brow  = chunk >> 6;              // 0..63  (n - nbase)
            int bcol  = chunk & 63;              // 16B unit within row
            unsigned      laddr = lbase + (unsigned)(brow * (BSTRIDE * 2) + bcol * 16);
            unsigned long long gaddr =
                (unsigned long long)(const void*)(gsrc + (size_t)chunk * 8);
            asm volatile("global_load_async_to_lds_b128 %0, %1, off"
                         :: "v"(laddr), "v"(gaddr) : "memory");
        }
#if __has_builtin(__builtin_amdgcn_s_wait_asynccnt)
        __builtin_amdgcn_s_wait_asynccnt(0);
#else
        asm volatile("s_wait_asynccnt 0x0" ::: "memory");
#endif
        __syncthreads();
    }

    // ---- Phase 2: per-wave 16x64 tile --------------------------------------
    const int wave  = tid >> 5;
    const int lane  = tid & 31;
    const int mtile = blockIdx.y * 8 + wave;
    if (mtile < Mtiles) {        // wave-uniform: EXEC stays all-ones for WMMA
        const int l16 = lane & 15;
        const int h   = lane >> 4;

        const bf16_t* ap = A + (size_t)(mtile * 16 + l16) * KDIM + h * 16;
        const bf16_t* lp0 = &Bs[(0 * 16 + l16) * BSTRIDE + h * 16];
        const bf16_t* lp1 = &Bs[(1 * 16 + l16) * BSTRIDE + h * 16];
        const bf16_t* lp2 = &Bs[(2 * 16 + l16) * BSTRIDE + h * 16];
        const bf16_t* lp3 = &Bs[(3 * 16 + l16) * BSTRIDE + h * 16];

        v8f c0 = {}, c1 = {}, c2 = {}, c3 = {};

        #pragma unroll 4
        for (int k0 = 0; k0 < KDIM; k0 += 32) {
            v8bf alo = *(const v8bf*)(ap + k0);
            v8bf ahi = *(const v8bf*)(ap + k0 + 8);
            if (k0 + 32 < KDIM)
                __builtin_prefetch(ap + k0 + 32, 0, 1);   // stream-ahead on A
            v16bf a = __builtin_shufflevector(alo, ahi,
                          0,1,2,3,4,5,6,7,8,9,10,11,12,13,14,15);
            v8bf b0l = *(const v8bf*)(lp0 + k0);
            v8bf b0h = *(const v8bf*)(lp0 + k0 + 8);
            v8bf b1l = *(const v8bf*)(lp1 + k0);
            v8bf b1h = *(const v8bf*)(lp1 + k0 + 8);
            v8bf b2l = *(const v8bf*)(lp2 + k0);
            v8bf b2h = *(const v8bf*)(lp2 + k0 + 8);
            v8bf b3l = *(const v8bf*)(lp3 + k0);
            v8bf b3h = *(const v8bf*)(lp3 + k0 + 8);
            v16bf b0 = __builtin_shufflevector(b0l, b0h,
                           0,1,2,3,4,5,6,7,8,9,10,11,12,13,14,15);
            v16bf b1 = __builtin_shufflevector(b1l, b1h,
                           0,1,2,3,4,5,6,7,8,9,10,11,12,13,14,15);
            v16bf b2 = __builtin_shufflevector(b2l, b2h,
                           0,1,2,3,4,5,6,7,8,9,10,11,12,13,14,15);
            v16bf b3 = __builtin_shufflevector(b3l, b3h,
                           0,1,2,3,4,5,6,7,8,9,10,11,12,13,14,15);
            c0 = __builtin_amdgcn_wmma_f32_16x16x32_bf16(false, a, false, b0,
                                                         (short)0, c0, false, false);
            c1 = __builtin_amdgcn_wmma_f32_16x16x32_bf16(false, a, false, b1,
                                                         (short)0, c1, false, false);
            c2 = __builtin_amdgcn_wmma_f32_16x16x32_bf16(false, a, false, b2,
                                                         (short)0, c2, false, false);
            c3 = __builtin_amdgcn_wmma_f32_16x16x32_bf16(false, a, false, b3,
                                                         (short)0, c3, false, false);
        }

        const int col = nbase + l16;
        const float bb0 = bias[col];
        const float bb1 = bias[col + 16];
        const float bb2 = bias[col + 32];
        const float bb3 = bias[col + 48];
        float* crow = C + (size_t)(mtile * 16 + h * 8) * N + col;
        #pragma unroll
        for (int r = 0; r < 8; ++r) {
            crow[0]  = c0[r] + bb0;
            crow[16] = c1[r] + bb1;
            crow[32] = c2[r] + bb2;
            crow[48] = c3[r] + bb3;
            crow += N;
        }
    }
}

// ---------------------------------------------------------------------------
// Attention (fp32, ~2 GFLOP total — VALU is fine). One 128-thread block per
// (b, q). Prefix-causal mask: visible iff k < 50 or k <= q.
// ---------------------------------------------------------------------------
__global__ __launch_bounds__(128) void attn_kernel(
    const float* __restrict__ Q, const float* __restrict__ Km,
    const float* __restrict__ V, float* __restrict__ out)
{
    __shared__ float sc[128];
    const int b = blockIdx.y;
    const int q = blockIdx.x;
    const int t = threadIdx.x;
    const float* qr = Q + ((size_t)b * SEQ + q) * DMODEL;

    float sim = -1e30f;
    if (t < SEQ) {
        if (t < IMG_LEN || t <= q) {
            const float* kr = Km + ((size_t)b * SEQ + t) * DMODEL;
            float acc = 0.f;
            #pragma unroll 4
            for (int i = 0; i < DMODEL; ++i) acc = fmaf(qr[i], kr[i], acc);
            sim = acc * 0.04419417382415922f;   // 1/sqrt(512)
        } else {
            sim = -1e9f;                        // matches reference mask value
        }
    }
    sc[t] = sim;
    __syncthreads();

    float mx = -1e30f;
    for (int j = 0; j < SEQ; ++j) mx = fmaxf(mx, sc[j]);
    float p = (t < SEQ) ? expf(sc[t] - mx) : 0.f;
    __syncthreads();
    sc[t] = p;
    __syncthreads();

    float ssum = 0.f;
    for (int j = 0; j < SEQ; ++j) ssum += sc[j];
    const float inv = 1.0f / ssum;

    #pragma unroll
    for (int c = 0; c < 4; ++c) {
        int d = t + c * 128;
        float acc = 0.f;
        for (int k = 0; k < SEQ; ++k)
            acc = fmaf(sc[k], V[((size_t)b * SEQ + k) * DMODEL + d], acc);
        out[((size_t)b * SEQ + q) * DMODEL + d] = acc * inv;
    }
}

// ---------------------------------------------------------------------------
// Gather the 76 text rows per batch into a dense [2432,512] bf16 matrix
// (2432 = 152 exact 16-row WMMA tiles) for the vocab GEMM.
// ---------------------------------------------------------------------------
__global__ __launch_bounds__(256) void gather_txt(
    const bf16_t* __restrict__ Xb, bf16_t* __restrict__ Xtb)
{
    int idx = blockIdx.x * 256 + threadIdx.x;   // TROWS*DMODEL
    if (idx >= TROWS * DMODEL) return;
    int r = idx >> 9;
    int d = idx & (DMODEL - 1);
    int b = r / TXT_LEN;
    int t = r - b * TXT_LEN;
    Xtb[idx] = Xb[((size_t)b * SEQ + IMG_LEN + t) * DMODEL + d];
}

// ---------------------------------------------------------------------------
// Orchestration. Workspace budget ~96 MB (d_out itself is 480 MB).
// ---------------------------------------------------------------------------
extern "C" void kernel_launch(void* const* d_in, const int* in_sizes, int n_in,
                              void* d_out, int out_size, void* d_ws, size_t ws_size,
                              hipStream_t stream)
{
    const float* img  = (const float*)d_in[0];
    const int*   ids  = (const int*)  d_in[1];
    const float* tok  = (const float*)d_in[2];
    const float* pos  = (const float*)d_in[3];
    const float* ff0w = (const float*)d_in[4];
    const float* ff0b = (const float*)d_in[5];
    const float* wq   = (const float*)d_in[6];
    const float* wqb  = (const float*)d_in[7];
    const float* wk   = (const float*)d_in[8];
    const float* wkb  = (const float*)d_in[9];
    const float* wv   = (const float*)d_in[10];
    const float* wvb  = (const float*)d_in[11];
    const float* lng  = (const float*)d_in[12];
    const float* lnb  = (const float*)d_in[13];
    const float* ff1w = (const float*)d_in[14];
    const float* ff1b = (const float*)d_in[15];
    const float* ff2w = (const float*)d_in[16];
    const float* ff2b = (const float*)d_in[17];
    float* out = (float*)d_out;

    char* ws = (char*)d_ws;
    auto take = [&ws](size_t bytes) {
        char* p = ws;
        ws += (bytes + 255) & ~(size_t)255;
        return p;
    };
    float*  X    = (float*)  take((size_t)ROWS  * DMODEL * 4);
    bf16_t* Xb   = (bf16_t*) take((size_t)ROWS  * DMODEL * 2);
    float*  Qb   = (float*)  take((size_t)ROWS  * DMODEL * 4);
    float*  Kb   = (float*)  take((size_t)ROWS  * DMODEL * 4);
    float*  Vb   = (float*)  take((size_t)ROWS  * DMODEL * 4);
    float*  Tb   = (float*)  take((size_t)ROWS  * DMODEL * 4);  // raw/attn out
    bf16_t* WqT  = (bf16_t*) take((size_t)DMODEL * DMODEL * 2);
    bf16_t* WkT  = (bf16_t*) take((size_t)DMODEL * DMODEL * 2);
    bf16_t* WvT  = (bf16_t*) take((size_t)DMODEL * DMODEL * 2);
    bf16_t* F1T  = (bf16_t*) take((size_t)DMODEL * DMODEL * 2);
    bf16_t* F2T  = (bf16_t*) take((size_t)VOCAB  * DMODEL * 2);
    bf16_t* Xtb  = (bf16_t*) take((size_t)TROWS  * DMODEL * 2);
    float*  Fb   = Qb;  // FF1 output reuses Q (dead after attention)

    // --- weight prep: transpose + bf16 convert -----------------------------
    {
        int n512 = (DMODEL * DMODEL + 255) / 256;
        transpose_to_bf16<<<n512, 256, 0, stream>>>(wq,  WqT, DMODEL, DMODEL);
        transpose_to_bf16<<<n512, 256, 0, stream>>>(wk,  WkT, DMODEL, DMODEL);
        transpose_to_bf16<<<n512, 256, 0, stream>>>(wv,  WvT, DMODEL, DMODEL);
        transpose_to_bf16<<<n512, 256, 0, stream>>>(ff1w, F1T, DMODEL, DMODEL);
        size_t e2 = (size_t)DMODEL * VOCAB;
        transpose_to_bf16<<<(unsigned)((e2 + 255) / 256), 256, 0, stream>>>(
            ff2w, F2T, DMODEL, VOCAB);
    }

    // --- embed + first LN --------------------------------------------------
    embed_kernel<<<(ROWS * DMODEL + 255) / 256, 256, 0, stream>>>(
        img, ids, tok, pos, ff0w, ff0b, Tb);
    ln_kernel<<<(ROWS + 7) / 8, 256, 0, stream>>>(
        Tb, nullptr, lng, lnb, X, Xb, ROWS);

    // --- 2 layers, shared params ------------------------------------------
    const int Mtiles = ROWS / 16;                  // 252
    dim3 gemm_grid(DMODEL / 64, (Mtiles + 7) / 8); // (8, 32)
    dim3 attn_grid(SEQ, BATCH);

    for (int layer = 0; layer < 2; ++layer) {
        gemm_bf16_wmma<<<gemm_grid, 256, 0, stream>>>(
            Xb, WqT, wqb, Qb, Mtiles, DMODEL);
        gemm_bf16_wmma<<<gemm_grid, 256, 0, stream>>>(
            Xb, WkT, wkb, Kb, Mtiles, DMODEL);
        gemm_bf16_wmma<<<gemm_grid, 256, 0, stream>>>(
            Xb, WvT, wvb, Vb, Mtiles, DMODEL);
        attn_kernel<<<attn_grid, 128, 0, stream>>>(Qb, Kb, Vb, Tb);
        ln_kernel<<<(ROWS + 7) / 8, 256, 0, stream>>>(      // x = LN(attn + res1)
            Tb, X, lng, lnb, X, Xb, ROWS);
        gemm_bf16_wmma<<<gemm_grid, 256, 0, stream>>>(
            Xb, F1T, ff1b, Fb, Mtiles, DMODEL);
        ln_kernel<<<(ROWS + 7) / 8, 256, 0, stream>>>(      // x = LN(ff1 + res2)
            Fb, X, lng, lnb, X, Xb, ROWS);
    }

    // --- vocab projection (the 123 GFLOP GEMM) -----------------------------
    gather_txt<<<(TROWS * DMODEL + 255) / 256, 256, 0, stream>>>(Xb, Xtb);
    dim3 logit_grid(VOCAB / 64, (TROWS / 16 + 7) / 8);      // (772, 19)
    gemm_bf16_wmma<<<logit_grid, 256, 0, stream>>>(
        Xtb, F2T, ff2b, out, TROWS / 16, VOCAB);
}
